// MultiHeadAttention_3496103379651
// MI455X (gfx1250) — matrix operand order repaired and checked by reading, not compiled, not run on hardware
//
#include <hip/hip_runtime.h>
#include <hip/hip_bf16.h>

#define EMBED 1024
#define HEADS 16
#define HDIM  64
#define NB    4
#define SEQ   2048

typedef __attribute__((ext_vector_type(16))) _Float16 v16h;
typedef __attribute__((ext_vector_type(8)))  _Float16 v8h;
typedef __attribute__((ext_vector_type(8)))  float    v8f;
typedef __attribute__((ext_vector_type(4)))  unsigned int u32v4;
typedef __attribute__((ext_vector_type(8)))  int i32v8;
typedef __attribute__((ext_vector_type(4)))  int i32v4;

// ---------------------------------------------------------------------------
// WMMA wrapper: D = A*B + C  (16x16x32, f16 in, f32 accumulate)
// ---------------------------------------------------------------------------
static __device__ __forceinline__ v8f wmma_f16(v16h a, v16h b, v8f c) {
    return __builtin_amdgcn_wmma_f32_16x16x32_f16(
        /*neg_a=*/false, a, /*neg_b=*/false, b,
        /*c_mod=*/(short)0, c, /*reuse_a=*/false, /*reuse_b=*/false);
}

// ---------------------------------------------------------------------------
// CDNA5 async copy: 16 bytes global -> LDS per lane, tracked by ASYNCcnt.
// ---------------------------------------------------------------------------
static __device__ __forceinline__ void async_copy16(unsigned lds_byte_addr,
                                                    const void* gaddr) {
    asm volatile("global_load_async_to_lds_b128 %0, %1, off"
                 :: "v"(lds_byte_addr), "v"(gaddr) : "memory");
}
static __device__ __forceinline__ void wait_async0() {
    asm volatile("s_wait_asynccnt 0x0" ::: "memory");
}
static __device__ __forceinline__ unsigned lds_addr(const void* p) {
    return (unsigned)(unsigned long long)p;   // low 32 bits = LDS byte offset
}

// ---------------------------------------------------------------------------
// CDNA5 Tensor Data Mover: 2D tile (16-bit elements) global -> LDS.
// D# built per ISA 8.3/8.4: group0 {count=1, lds_addr, global_addr, type=2},
// group1 {data_size=2B, tensor_dim0/1 == tile dims (in-bounds), tile_dim0/1,
// 48-bit dim0 stride}.  Tile packed contiguously row-major into LDS.
// This toolchain exposes the 6-arg builtin:
// (u32x4 g0, i32x8 g1, i32x4, i32x4, i32x8, i32 cpol).
// ---------------------------------------------------------------------------
static __device__ __forceinline__ void tdm_load_2d(
    unsigned lds_byte, const void* gaddr,
    unsigned tile_d0 /*elems per row*/, unsigned tile_d1 /*rows*/,
    unsigned row_stride /*elems*/)
{
    unsigned long long ga = (unsigned long long)gaddr;
    u32v4 g0;
    g0[0] = 1u;                                            // count=1 (valid), user
    g0[1] = lds_byte;                                      // lds_addr [63:32]
    g0[2] = (unsigned)(ga & 0xFFFFFFFFu);                  // global_addr[31:0]
    g0[3] = (unsigned)((ga >> 32) & 0x1FFFFFFu) | (2u << 30); // addr[56:32] | type=2
    i32v8 g1;
    g1[0] = (int)(1u << 16);                               // data_size=1 (2 bytes)
    g1[1] = (int)((tile_d0 & 0xFFFFu) << 16);              // tensor_dim0[15:0] @ [63:48]
    g1[2] = (int)(((tile_d0 >> 16) & 0xFFFFu) |            // tensor_dim0[31:16]
                  ((tile_d1 & 0xFFFFu) << 16));            // tensor_dim1[15:0]
    g1[3] = (int)(((tile_d1 >> 16) & 0xFFFFu) |            // tensor_dim1[31:16]
                  ((tile_d0 & 0xFFFFu) << 16));            // tile_dim0 @ [127:112]
    g1[4] = (int)(tile_d1 & 0xFFFFu);                      // tile_dim1; tile_dim2=0
    g1[5] = (int)row_stride;                               // tensor_dim0_stride[31:0]
    g1[6] = 0;                                             // stride hi, dim1_stride lo
    g1[7] = 0;
    i32v4 z4 = {0, 0, 0, 0};
    i32v8 z8 = {0, 0, 0, 0, 0, 0, 0, 0};
    __builtin_amdgcn_tensor_load_to_lds(g0, g1, z4, z4, z8, 0);
}
static __device__ __forceinline__ void wait_tensor0() {
    __builtin_amdgcn_s_wait_tensorcnt(0);
}

// ---------------------------------------------------------------------------
// Fragment loaders.
// A layout (16x32, MxK): lane<16 holds row M=lane, K = {0..7, 16..23};
//                        lane>=16 holds row M=lane-16, K = {8..15, 24..31}.
// B layout (32x16, KxN): lane holds col N=lane&15, K = 8*(lane>>4) + {0..7, 16..23}.
// Source is row-major MxK for A, row-major B^T (NxK) for B.
// ---------------------------------------------------------------------------
static __device__ __forceinline__ v16h load_fragA_f16(const _Float16* __restrict__ p, int ld) {
    const int lane = threadIdx.x & 31;
    const int r  = lane & 15;
    const int kb = (lane >> 4) << 3;          // 0 or 8
    const _Float16* q = p + r * ld + kb;
    v8h lo = *(const v8h*)(q);                // K = kb .. kb+7
    v8h hi = *(const v8h*)(q + 16);           // K = kb+16 .. kb+23
    v16h out;
#pragma unroll
    for (int i = 0; i < 8; ++i) { out[i] = lo[i]; out[i + 8] = hi[i]; }
    return out;
}

static __device__ __forceinline__ v16h load_fragA_f32cvt(const float* __restrict__ p, int ld) {
    const int lane = threadIdx.x & 31;
    const int r  = lane & 15;
    const int kb = (lane >> 4) << 3;
    const float* q = p + r * ld + kb;
    v16h out;
#pragma unroll
    for (int i = 0; i < 8; ++i) { out[i] = (_Float16)q[i]; out[i + 8] = (_Float16)q[i + 16]; }
    return out;
}

static __device__ __forceinline__ v16h load_fragB_f16(const _Float16* __restrict__ p, int ld) {
    const int lane = threadIdx.x & 31;
    const int n  = lane & 15;
    const int kb = (lane >> 4) << 3;          // 0 or 8
    const _Float16* q = p + n * ld + kb;
    v8h lo = *(const v8h*)(q);
    v8h hi = *(const v8h*)(q + 16);
    v16h out;
#pragma unroll
    for (int i = 0; i < 8; ++i) { out[i] = lo[i]; out[i + 8] = hi[i]; }
    return out;
}

static __device__ __forceinline__ v16h load_fragB_f32cvt(const float* __restrict__ p, int ld) {
    const int lane = threadIdx.x & 31;
    const int n  = lane & 15;
    const int kb = (lane >> 4) << 3;
    const float* q = p + n * ld + kb;
    v16h out;
#pragma unroll
    for (int i = 0; i < 8; ++i) { out[i] = (_Float16)q[i]; out[i + 8] = (_Float16)q[i + 16]; }
    return out;
}

// ---------------------------------------------------------------------------
// Accumulator (C/D) stores. Element (m,n): lane = (n&15) + 16*(m/8),
// vgpr i holds row m = 8*(lane>>4) + i.
// ---------------------------------------------------------------------------
static __device__ __forceinline__ void store_tile_f16(_Float16* __restrict__ p, int ld, v8f c) {
    const int lane = threadIdx.x & 31;
    const int col = lane & 15;
    const int rb  = (lane >> 4) << 3;
#pragma unroll
    for (int i = 0; i < 8; ++i) p[(rb + i) * ld + col] = (_Float16)c[i];
}

// Transposed store: element (m,n) -> p[n*ldT + m]  (vectorizes to one b128)
static __device__ __forceinline__ void store_tile_f16_T(_Float16* __restrict__ p, int ldT, v8f c) {
    const int lane = threadIdx.x & 31;
    const int col = lane & 15;
    const int rb  = (lane >> 4) << 3;
    v8h v;
#pragma unroll
    for (int i = 0; i < 8; ++i) v[i] = (_Float16)c[i];
    *(v8h*)(p + col * ldT + rb) = v;
}

// ===========================================================================
// Kernel 0: convert Wo (fp32) -> f16
// ===========================================================================
__global__ void cvt_f32_f16_kernel(const float* __restrict__ in,
                                   _Float16* __restrict__ out, int n) {
    int i = blockIdx.x * blockDim.x + threadIdx.x;
    if (i < n) out[i] = (_Float16)in[i];
}

// ===========================================================================
// Kernel 1: QKV projection.  One block = (n, h, 128 rows of L); 8 waves,
// each wave computes a 16x64 tile of Q, K, V via WMMA.
// Q, K row-major f16 [N,H,L,64]; V transposed f16 [N,H,64,L].
// ===========================================================================
__global__ __launch_bounds__(256) void qkv_proj_kernel(
    const float* __restrict__ x,
    const float* __restrict__ Wq, const float* __restrict__ Wk,
    const float* __restrict__ Wv,
    _Float16* __restrict__ Qb, _Float16* __restrict__ Kb,
    _Float16* __restrict__ Vt)
{
    const int w = threadIdx.x >> 5;
    int b = blockIdx.x;
    const int TPL = SEQ / 128;
    const int lt = b % TPL; b /= TPL;
    const int h  = b % HEADS; b /= HEADS;
    const int n  = b;
    const int l0 = lt * 128 + w * 16;

    const float* xp = x + ((size_t)(n * SEQ + l0)) * EMBED + h * HDIM;
    const v16h a0 = load_fragA_f32cvt(xp, EMBED);        // k = 0..31
    const v16h a1 = load_fragA_f32cvt(xp + 32, EMBED);   // k = 32..63

    const size_t hoff = ((size_t)(n * HEADS + h)) * SEQ * HDIM;
    _Float16* qo = Qb + hoff + (size_t)l0 * HDIM;
    _Float16* ko = Kb + hoff + (size_t)l0 * HDIM;
    _Float16* vo = Vt + hoff + l0;                       // [64][SEQ] origin at col l0

#pragma unroll
    for (int nt = 0; nt < 4; ++nt) {
        // Q
        v8f c = {};
        c = wmma_f16(a0, load_fragB_f32cvt(Wq + nt * 16 * HDIM, HDIM), c);
        c = wmma_f16(a1, load_fragB_f32cvt(Wq + nt * 16 * HDIM + 32, HDIM), c);
        store_tile_f16(qo + nt * 16, HDIM, c);
        // K
        v8f ck = {};
        ck = wmma_f16(a0, load_fragB_f32cvt(Wk + nt * 16 * HDIM, HDIM), ck);
        ck = wmma_f16(a1, load_fragB_f32cvt(Wk + nt * 16 * HDIM + 32, HDIM), ck);
        store_tile_f16(ko + nt * 16, HDIM, ck);
        // V (transposed store for PV GEMM B-operand)
        v8f cv = {};
        cv = wmma_f16(a0, load_fragB_f32cvt(Wv + nt * 16 * HDIM, HDIM), cv);
        cv = wmma_f16(a1, load_fragB_f32cvt(Wv + nt * 16 * HDIM + 32, HDIM), cv);
        store_tile_f16_T(vo + (size_t)(nt * 16) * SEQ, SEQ, cv);
    }
}

// ===========================================================================
// Kernel 2: flash attention (FA-2), fused QK^T -> softmax -> PV.
// Block = (n, h, 128 query rows), 8 waves x 16 rows.
// K/V key-blocks staged once per block into LDS by the Tensor Data Mover:
// wave 0 issues two TENSOR_LOAD_TO_LDS descriptors per block (K 32x64,
// V^T 64x32 strided rows), waits TENSORcnt, double-buffered so the DMA of
// block i+1 overlaps the WMMA work on block i.
// ===========================================================================
__global__ __launch_bounds__(256) void flash_attn_kernel(
    const _Float16* __restrict__ Qb, const _Float16* __restrict__ Kb,
    const _Float16* __restrict__ Vt, _Float16* __restrict__ Ob)
{
    __shared__ __align__(16) _Float16 kbuf[2][32 * HDIM];  // 32 keys x 64 (row-major)
    __shared__ __align__(16) _Float16 vbuf[2][HDIM * 32];  // 64 d    x 32 keys (V^T)
    __shared__ __align__(16) _Float16 pbuf[8][16 * 32];    // per-wave P staging

    const int w    = threadIdx.x >> 5;
    const int lane = threadIdx.x & 31;
    const int tid  = threadIdx.x;
    int b = blockIdx.x;
    const int TPL = SEQ / 128;
    const int lt = b % TPL; b /= TPL;
    const int h  = b % HEADS; b /= HEADS;
    const int n  = b;
    const int l0 = lt * 128 + w * 16;

    const size_t hoff = ((size_t)(n * HEADS + h)) * SEQ * HDIM;
    const _Float16* Qp = Qb + hoff + (size_t)l0 * HDIM;
    const _Float16* Kp = Kb + hoff;
    const _Float16* Vp = Vt + hoff;

    // TDM staging of one 32-key block (8 KB): issued by wave 0 only.
    auto stage = [&](int bi, int kb) {
        tdm_load_2d(lds_addr(&kbuf[bi][0]), Kp + (size_t)kb * HDIM,
                    /*tile_d0=*/HDIM, /*tile_d1=*/32, /*row_stride=*/HDIM);
        tdm_load_2d(lds_addr(&vbuf[bi][0]), Vp + kb,
                    /*tile_d0=*/32, /*tile_d1=*/HDIM, /*row_stride=*/SEQ);
    };

    const v16h a0 = load_fragA_f16(Qp, HDIM);
    const v16h a1 = load_fragA_f16(Qp + 32, HDIM);

    v8f o0 = {}, o1 = {}, o2 = {}, o3 = {};
    float mrow[8], lrow[8];
#pragma unroll
    for (int i = 0; i < 8; ++i) { mrow[i] = -1e30f; lrow[i] = 0.0f; }

    // scale = 1/sqrt(EMBED) folded with log2(e) for exp2-domain softmax
    const float SL = (1.0f / 32.0f) * 1.44269504088896340736f;

    _Float16* lp = &pbuf[w][0];
    const int col = lane & 15;
    const int rb  = (lane >> 4) << 3;

    if (tid < 32) { stage(0, 0); wait_tensor0(); }
    __syncthreads();
    int buf = 0;

#pragma unroll 1
    for (int kb = 0; kb < SEQ; kb += 32) {
        if (tid < 32 && kb + 32 < SEQ) stage(buf ^ 1, kb + 32);  // overlap DMA

        const _Float16* kt = &kbuf[buf][0];
        const _Float16* vt = &vbuf[buf][0];

        // ---- S = Q K^T (16 x 32 tile) -----------------------------------
        v8f s0 = {}, s1 = {};
        s0 = wmma_f16(a0, load_fragB_f16(kt, HDIM), s0);
        s0 = wmma_f16(a1, load_fragB_f16(kt + 32, HDIM), s0);
        s1 = wmma_f16(a0, load_fragB_f16(kt + 16 * HDIM, HDIM), s1);
        s1 = wmma_f16(a1, load_fragB_f16(kt + 16 * HDIM + 32, HDIM), s1);

        // ---- online softmax (row = 8*(lane>>4)+i, spans 16 lanes) -------
        float alpha[8];
#pragma unroll
        for (int i = 0; i < 8; ++i) {
            float t0 = s0[i] * SL, t1 = s1[i] * SL;
            float mx = fmaxf(t0, t1);
            mx = fmaxf(mx, __shfl_xor(mx, 1, 32));
            mx = fmaxf(mx, __shfl_xor(mx, 2, 32));
            mx = fmaxf(mx, __shfl_xor(mx, 4, 32));
            mx = fmaxf(mx, __shfl_xor(mx, 8, 32));
            const float mnew = fmaxf(mrow[i], mx);
            const float p0 = __builtin_exp2f(t0 - mnew);
            const float p1 = __builtin_exp2f(t1 - mnew);
            const float a  = __builtin_exp2f(mrow[i] - mnew);
            mrow[i] = mnew;
            float rs = p0 + p1;
            rs += __shfl_xor(rs, 1, 32);
            rs += __shfl_xor(rs, 2, 32);
            rs += __shfl_xor(rs, 4, 32);
            rs += __shfl_xor(rs, 8, 32);
            lrow[i] = lrow[i] * a + rs;
            alpha[i] = a;
            s0[i] = p0; s1[i] = p1;
        }
#pragma unroll
        for (int i = 0; i < 8; ++i) {
            o0[i] *= alpha[i]; o1[i] *= alpha[i];
            o2[i] *= alpha[i]; o3[i] *= alpha[i];
        }

        // ---- P: D-layout -> LDS -> A-layout -----------------------------
#pragma unroll
        for (int i = 0; i < 8; ++i) {
            lp[(rb + i) * 32 + col]      = (_Float16)s0[i];
            lp[(rb + i) * 32 + 16 + col] = (_Float16)s1[i];
        }
        asm volatile("s_wait_dscnt 0x0" ::: "memory");  // wave-internal LDS RAW
        const v16h ap = load_fragA_f16(lp, 32);

        // ---- O += P V  (B operand from LDS-resident V^T tile) -----------
        o0 = wmma_f16(ap, load_fragB_f16(vt + 0 * 16 * 32, 32), o0);
        o1 = wmma_f16(ap, load_fragB_f16(vt + 1 * 16 * 32, 32), o1);
        o2 = wmma_f16(ap, load_fragB_f16(vt + 2 * 16 * 32, 32), o2);
        o3 = wmma_f16(ap, load_fragB_f16(vt + 3 * 16 * 32, 32), o3);

        if (tid < 32) wait_tensor0();   // my DMA of buf^1 has landed
        __syncthreads();                // everyone done reading buf; buf^1 ready
        buf ^= 1;
    }

    // ---- finalize: O /= l, store into [N, L, H*64] f16 -------------------
    float rinv[8];
#pragma unroll
    for (int i = 0; i < 8; ++i) rinv[i] = 1.0f / lrow[i];
#pragma unroll
    for (int i = 0; i < 8; ++i) {
        o0[i] *= rinv[i]; o1[i] *= rinv[i];
        o2[i] *= rinv[i]; o3[i] *= rinv[i];
    }
    _Float16* op = Ob + ((size_t)(n * SEQ + l0)) * EMBED + h * HDIM;
    store_tile_f16(op + 0,  EMBED, o0);
    store_tile_f16(op + 16, EMBED, o1);
    store_tile_f16(op + 32, EMBED, o2);
    store_tile_f16(op + 48, EMBED, o3);
}

// ===========================================================================
// Kernel 3: output projection  Y = O * Wo^T + bo   (fp32 out)
// Block = 128 rows x 64 cols; wave = 16 rows x 64 cols.
// Shared 64x32 Wo tile per k-step async-staged into LDS (ASYNCcnt path),
// double-buffered; next A tile prefetched (global_prefetch).
// ===========================================================================
__global__ __launch_bounds__(256) void out_proj_kernel(
    const _Float16* __restrict__ Ob, const _Float16* __restrict__ Wo16,
    const float* __restrict__ bo, float* __restrict__ y)
{
    __shared__ __align__(16) _Float16 wbuf[2][64 * 32];   // 64 n-rows x 32 k

    const int w    = threadIdx.x >> 5;
    const int lane = threadIdx.x & 31;
    const int tid  = threadIdx.x;
    const int m0 = blockIdx.x * 128 + w * 16;
    const int n0 = blockIdx.y * 64;

    const int wr = tid >> 2, wc = (tid & 3) << 3;   // 64 rows x 4 chunks of 8 f16
    auto stage = [&](int bi, int k) {
        async_copy16(lds_addr(&wbuf[bi][wr * 32 + wc]),
                     Wo16 + (size_t)(n0 + wr) * EMBED + k + wc);
    };

    const _Float16* ap = Ob + (size_t)m0 * EMBED;
    v8f c0 = {}, c1 = {}, c2 = {}, c3 = {};

    stage(0, 0);
    wait_async0();
    __syncthreads();
    int buf = 0;

#pragma unroll 1
    for (int k = 0; k < EMBED; k += 32) {
        if (k + 32 < EMBED) {
            stage(buf ^ 1, k + 32);
            __builtin_prefetch(ap + k + 32, 0, 0);    // global_prefetch_b8
        }

        const v16h a = load_fragA_f16(ap + k, EMBED);
        const _Float16* wt = &wbuf[buf][0];
        c0 = wmma_f16(a, load_fragB_f16(wt + 0 * 16 * 32, 32), c0);
        c1 = wmma_f16(a, load_fragB_f16(wt + 1 * 16 * 32, 32), c1);
        c2 = wmma_f16(a, load_fragB_f16(wt + 2 * 16 * 32, 32), c2);
        c3 = wmma_f16(a, load_fragB_f16(wt + 3 * 16 * 32, 32), c3);

        wait_async0();
        __syncthreads();
        buf ^= 1;
    }

    const int col = lane & 15;
    const int rb  = (lane >> 4) << 3;
    const v8f acc[4] = {c0, c1, c2, c3};
#pragma unroll
    for (int t = 0; t < 4; ++t) {
        const int nc = n0 + t * 16 + col;
        const float bias = bo[nc];
#pragma unroll
        for (int i = 0; i < 8; ++i)
            y[(size_t)(m0 + rb + i) * EMBED + nc] = acc[t][i] + bias;
    }
}

// ===========================================================================
extern "C" void kernel_launch(void* const* d_in, const int* in_sizes, int n_in,
                              void* d_out, int out_size, void* d_ws, size_t ws_size,
                              hipStream_t stream) {
    const float* x  = (const float*)d_in[0];
    const float* Wq = (const float*)d_in[1];
    const float* Wk = (const float*)d_in[2];
    const float* Wv = (const float*)d_in[3];
    const float* Wo = (const float*)d_in[4];
    const float* bo = (const float*)d_in[5];
    float* y = (float*)d_out;

    // workspace layout (f16): Q 16MB | K 16MB | Vt 16MB | O 16MB | Wo16 2MB
    char* ws = (char*)d_ws;
    const size_t szQ = (size_t)NB * HEADS * SEQ * HDIM * sizeof(_Float16);
    _Float16* Qb   = (_Float16*)(ws);
    _Float16* Kb   = (_Float16*)(ws + szQ);
    _Float16* Vt   = (_Float16*)(ws + 2 * szQ);
    _Float16* Ob   = (_Float16*)(ws + 3 * szQ);
    _Float16* Wo16 = (_Float16*)(ws + 4 * szQ);

    cvt_f32_f16_kernel<<<(EMBED * EMBED + 255) / 256, 256, 0, stream>>>(
        Wo, Wo16, EMBED * EMBED);

    const int nblk = NB * HEADS * (SEQ / 128);
    qkv_proj_kernel<<<nblk, 256, 0, stream>>>(x, Wq, Wk, Wv, Qb, Kb, Vt);
    flash_attn_kernel<<<nblk, 256, 0, stream>>>(Qb, Kb, Vt, Ob);

    dim3 g(NB * SEQ / 128, EMBED / 64);
    out_proj_kernel<<<g, 256, 0, stream>>>(Ob, Wo16, bo, y);
}
